// GGNN_12223476925005
// MI455X (gfx1250) — matrix operand order repaired and checked by reading, not compiled, not run on hardware
//
#include <hip/hip_runtime.h>
#include <hip/hip_bf16.h>

// ---------------------------------------------------------------------------
// GGNN on MI455X (gfx1250). All three edge-MLP layers run on
// v_wmma_f32_16x16x32_f16; layer 1 is a K=96 GEMM over per-edge feature rows
// E = [h_i | h_j | J, b_i, b_j, 1 | 0-pad], with the bias folded into the
// "1" column so the C accumulator starts as inline 0.
// N=512 nodes, d=32, msg=32, H_MSG=64, 5 steps.
// ---------------------------------------------------------------------------

typedef __attribute__((ext_vector_type(16))) _Float16 v16h;
typedef __attribute__((ext_vector_type(8)))  _Float16 v8h;
typedef __attribute__((ext_vector_type(8)))  float    v8f;

#define NN 512
#define DD 32
#define HM 64
#define MM 32

#define ESTR 104   // E-row stride in halfs (96 + 8 pad, 208 B, 16B-aligned)
#define MSTR 72    // m1/m2 row stride in halfs (64 + 8 pad, 144 B)

#define LDS_M1_BYTES   (256 * MSTR * 2)                 // 36864
#define LDS_E_BYTES    (256 * ESTR * 2)                 // 53248
#define LDS_PART_BYTES (512 * 4)                        // 2048
#define LDS_TOTAL      (LDS_M1_BYTES + LDS_E_BYTES + LDS_PART_BYTES) // 92160

static __device__ __forceinline__ v16h cat8(v8h a, v8h b) {
    return __builtin_shufflevector(a, b, 0,1,2,3,4,5,6,7,8,9,10,11,12,13,14,15);
}

static __device__ __forceinline__ v8f wmma16(v16h a, v16h b, v8f c) {
    // D = A(16x32 f16) * B(32x16 f16) + C(16x16 f32)
    return __builtin_amdgcn_wmma_f32_16x16x32_f16(
        false, a, false, b, (short)0, c, false, false);
}

// A-fragment (16x32 f16) from an LDS tile with given row stride (halfs).
// Lane l: row = l&15; v0..v3 = K kc*32+(l>>4)*8+{0..7}, v4..v7 = +16.
static __device__ __forceinline__ v16h load_a_lds(const _Float16* base, int stride,
                                                  int lane, int kc) {
    int row = lane & 15;
    int kh  = lane >> 4;
    const _Float16* p = base + row * stride + kc * 32 + kh * 8;
    v8h lo = *(const v8h*)p;
    v8h hi = *(const v8h*)(p + 16);
    return cat8(lo, hi);
}

// B-fragment (32x16 f16) from row-major weight Wh[n][k], row stride in halfs.
// Lane l: col n = nt*16 + (l&15); v0..v3 = K kc*32+(l>>4)*8+{0..7}, v4..v7 = +16.
static __device__ __forceinline__ v16h load_b_row(const _Float16* Wh, int stride,
                                                  int lane, int nt, int kc) {
    int n  = nt * 16 + (lane & 15);
    int kh = lane >> 4;
    const _Float16* p = Wh + n * stride + kc * 32 + kh * 8;
    v8h lo = *(const v8h*)p;
    v8h hi = *(const v8h*)(p + 16);
    return cat8(lo, hi);
}

static __device__ __forceinline__ v8h cvt8(const float* p) {
    v8h r;
#pragma unroll
    for (int u = 0; u < 8; ++u) r[u] = (_Float16)p[u];
    return r;
}

// ---------------------------------------------------------------------------
// Prep: f16 weight copies. W1h is 64 x 96: cols 0..66 = mp_W1, col 67 = b1
// (bias column matching E's constant-1 column), cols 68..95 = 0.
// Also zero h0 and msg_sum.
// ---------------------------------------------------------------------------
__global__ __launch_bounds__(256)
void ggnn_prep(const float* __restrict__ W1, const float* __restrict__ b1,
               const float* __restrict__ W2, const float* __restrict__ W3,
               _Float16* __restrict__ W1h, _Float16* __restrict__ W2h,
               _Float16* __restrict__ W3h,
               float* __restrict__ h0, float* __restrict__ msg_sum) {
    int t = blockIdx.x * 256 + threadIdx.x;
    if (t < HM * 96) {
        int r = t / 96, c = t % 96;
        float v = (c < 67) ? W1[r * 67 + c] : (c == 67 ? b1[r] : 0.f);
        W1h[t] = (_Float16)v;
    }
    if (t < HM * HM) W2h[t] = (_Float16)W2[t];
    if (t < MM * HM) W3h[t] = (_Float16)W3[t];
    if (t < NN * DD) { h0[t] = 0.f; msg_sum[t] = 0.f; }
}

// ---------------------------------------------------------------------------
// Edge kernel: block = 16x16 (i,j) edge tile, 256 threads = 8 waves.
// E build -> WMMA layer1 (K=96, bias in-K) -> WMMA layer2 -> WMMA layer3
// -> msg reduction over i.
// ---------------------------------------------------------------------------
__global__ __launch_bounds__(256)
void ggnn_edge(const float* __restrict__ J, const float* __restrict__ b,
               const float* __restrict__ h,
               const _Float16* __restrict__ W1h,
               const _Float16* __restrict__ W2h, const float* __restrict__ b2,
               const _Float16* __restrict__ W3h, const float* __restrict__ b3,
               float* __restrict__ msg_sum) {
    extern __shared__ char lds_raw[];
    _Float16* sm1   = (_Float16*)lds_raw;                        // 256 x MSTR
    _Float16* sE    = (_Float16*)(lds_raw + LDS_M1_BYTES);       // 256 x ESTR
    float*    spart = (float*)(lds_raw + LDS_M1_BYTES + LDS_E_BYTES); // 16x32

    const int t    = threadIdx.x;
    const int wave = t >> 5;
    const int lane = t & 31;
    const int i0   = blockIdx.x * 16;
    const int j0   = blockIdx.y * 16;

    spart[t] = 0.f;
    spart[t + 256] = 0.f;
    __syncthreads();

    // ---- Build E row for edge e = t (i_l = t>>4, j_l = t&15) ----
    {
        const int il = t >> 4, jl = t & 15;
        const int ig = i0 + il, jg = j0 + jl;
        const float Jv = J[ig * NN + jg];
        const float bi = b[ig], bj = b[jg];
        const float* hi = h + ig * DD;
        const float* hj = h + jg * DD;
        _Float16* er = sE + t * ESTR;
#pragma unroll
        for (int q = 0; q < 4; ++q) *(v8h*)(er + q * 8)      = cvt8(hi + q * 8);
#pragma unroll
        for (int q = 0; q < 4; ++q) *(v8h*)(er + 32 + q * 8) = cvt8(hj + q * 8);
        // cols 64..95: J, b_i, b_j, 1 (bias column), then zeros
        v8h tail = {(_Float16)Jv, (_Float16)bi, (_Float16)bj,
                    (_Float16)1.f, (_Float16)0.f, (_Float16)0.f,
                    (_Float16)0.f, (_Float16)0.f};
        v8h zero = {(_Float16)0.f, (_Float16)0.f, (_Float16)0.f, (_Float16)0.f,
                    (_Float16)0.f, (_Float16)0.f, (_Float16)0.f, (_Float16)0.f};
        *(v8h*)(er + 64) = tail;
        *(v8h*)(er + 72) = zero;
        *(v8h*)(er + 80) = zero;
        *(v8h*)(er + 88) = zero;
    }
    // E rows [32w, 32w+32) are produced and consumed only by wave w:
    // per-wave DS ordering suffices, no block barrier needed.

    const int ncol  = lane & 15;        // C-fragment column within n-tile
    const int rbase = (lane >> 4) * 8;  // C-fragment row base

    // ---- Layer 1 via WMMA: m1 = relu(E @ W1h^T), K = 96, C starts at 0 ----
#pragma unroll
    for (int mi = 0; mi < 2; ++mi) {
        const int mt = 2 * wave + mi;
        const _Float16* eb = sE + mt * 16 * ESTR;
        v16h a0 = load_a_lds(eb, ESTR, lane, 0);
        v16h a1 = load_a_lds(eb, ESTR, lane, 1);
        v16h a2 = load_a_lds(eb, ESTR, lane, 2);
#pragma unroll
        for (int nt = 0; nt < 4; ++nt) {
            v8f c = {0.f, 0.f, 0.f, 0.f, 0.f, 0.f, 0.f, 0.f};  // inline-0 SRC2
            c = wmma16(a0, load_b_row(W1h, 96, lane, nt, 0), c);
            c = wmma16(a1, load_b_row(W1h, 96, lane, nt, 1), c);
            c = wmma16(a2, load_b_row(W1h, 96, lane, nt, 2), c);
#pragma unroll
            for (int r = 0; r < 8; ++r) {
                float v = c[r];
                sm1[(mt * 16 + rbase + r) * MSTR + nt * 16 + ncol] =
                    (_Float16)(v > 0.f ? v : 0.f);
            }
        }
    }

    // ---- Resident B fragments for W2 (4x2) and W3 (2x2) ----
    v16h bw2[4][2], bw3[2][2];
#pragma unroll
    for (int nt = 0; nt < 4; ++nt)
#pragma unroll
        for (int kc = 0; kc < 2; ++kc)
            bw2[nt][kc] = load_b_row(W2h, 64, lane, nt, kc);
#pragma unroll
    for (int nt = 0; nt < 2; ++nt)
#pragma unroll
        for (int kc = 0; kc < 2; ++kc)
            bw3[nt][kc] = load_b_row(W3h, 64, lane, nt, kc);

    // m2 tile aliases this wave's own (now dead) E slice -> wave-local reuse.
    _Float16* m2w = sE + wave * 32 * ESTR;

    float macc[2][8];
#pragma unroll
    for (int nt = 0; nt < 2; ++nt)
#pragma unroll
        for (int r = 0; r < 8; ++r) macc[nt][r] = 0.f;

    // ---- Layers 2+3 per wave over its two 16-edge M-tiles ----
#pragma unroll
    for (int mi = 0; mi < 2; ++mi) {
        const int mt = 2 * wave + mi;
        const _Float16* m1b = sm1 + mt * 16 * MSTR;
        v16h a0 = load_a_lds(m1b, MSTR, lane, 0);
        v16h a1 = load_a_lds(m1b, MSTR, lane, 1);

        // Layer 2: m2 = relu(m1 @ W2^T + b2)
#pragma unroll
        for (int nt = 0; nt < 4; ++nt) {
            float bb = b2[nt * 16 + ncol];
            v8f c = {bb, bb, bb, bb, bb, bb, bb, bb};
            c = wmma16(a0, bw2[nt][0], c);
            c = wmma16(a1, bw2[nt][1], c);
#pragma unroll
            for (int r = 0; r < 8; ++r) {
                float v = c[r];
                m2w[(rbase + r) * MSTR + nt * 16 + ncol] =
                    (_Float16)(v > 0.f ? v : 0.f);
            }
        }

        v16h a20 = load_a_lds(m2w, MSTR, lane, 0);
        v16h a21 = load_a_lds(m2w, MSTR, lane, 1);

        // Layer 3: msg = relu(m2 @ W3^T + b3); C row index == j_l, so summing
        // this wave's two M-tiles reduces over its i values.
#pragma unroll
        for (int nt = 0; nt < 2; ++nt) {
            float bb = b3[nt * 16 + ncol];
            v8f c = {bb, bb, bb, bb, bb, bb, bb, bb};
            c = wmma16(a20, bw3[nt][0], c);
            c = wmma16(a21, bw3[nt][1], c);
#pragma unroll
            for (int r = 0; r < 8; ++r) {
                float v = c[r];
                macc[nt][r] += (v > 0.f ? v : 0.f);
            }
        }
    }

    // ---- Cross-wave reduction in LDS, then one global atomic per value ----
#pragma unroll
    for (int nt = 0; nt < 2; ++nt)
#pragma unroll
        for (int r = 0; r < 8; ++r)
            atomicAdd(&spart[(rbase + r) * 32 + nt * 16 + ncol], macc[nt][r]);
    __syncthreads();

#pragma unroll
    for (int u = 0; u < 2; ++u) {
        int idx = t + u * 256;
        int jl = idx >> 5, ch = idx & 31;
        atomicAdd(&msg_sum[(j0 + jl) * MM + ch], spart[idx]);
    }
}

// ---------------------------------------------------------------------------
// GRU update; consumes msg_sum and re-zeros it (block-local, race-free).
// ---------------------------------------------------------------------------
__global__ __launch_bounds__(256)
void ggnn_gru(const float* __restrict__ h, float* __restrict__ msg_sum,
              const float* __restrict__ Wih, const float* __restrict__ Whh,
              const float* __restrict__ bih, const float* __restrict__ bhh,
              float* __restrict__ h_new) {
    int t = blockIdx.x * 256 + threadIdx.x;   // 512*32
    int n = t >> 5, k = t & 31;
    const float* hp = h + n * DD;
    const float* mp = msg_sum + n * MM;

    float gi[3], gh[3];
#pragma unroll
    for (int g = 0; g < 3; ++g) {
        const float* wr = Wih + (g * DD + k) * (DD + MM);
        float acc = bih[g * DD + k];
#pragma unroll
        for (int c = 0; c < DD; ++c) acc += wr[c] * hp[c];
#pragma unroll
        for (int c = 0; c < MM; ++c) acc += wr[DD + c] * mp[c];
        gi[g] = acc;
        const float* wh = Whh + (g * DD + k) * DD;
        float acc2 = bhh[g * DD + k];
#pragma unroll
        for (int c = 0; c < DD; ++c) acc2 += wh[c] * hp[c];
        gh[g] = acc2;
    }
    float r  = 1.f / (1.f + __expf(-(gi[0] + gh[0])));
    float z  = 1.f / (1.f + __expf(-(gi[1] + gh[1])));
    float ng = tanhf(gi[2] + r * gh[2]);
    float hn = (1.f - z) * ng + z * hp[k];

    __syncthreads();           // all 32 k-threads of each node are in-block
    msg_sum[n * MM + k] = 0.f; // ready for next step
    h_new[t] = hn;
}

// ---------------------------------------------------------------------------
// Readout MLP + sigmoid -> out (512 x 2 f32).
// ---------------------------------------------------------------------------
__global__ __launch_bounds__(64)
void ggnn_readout(const float* __restrict__ h,
                  const float* __restrict__ W1, const float* __restrict__ b1,
                  const float* __restrict__ W2, const float* __restrict__ b2,
                  const float* __restrict__ W3, const float* __restrict__ b3,
                  float* __restrict__ out) {
    int n = blockIdx.x * 64 + threadIdx.x;    // 512 nodes
    const float* hp = h + n * DD;
    float y1[64], y2[64];
#pragma unroll 4
    for (int o = 0; o < 64; ++o) {
        float a = b1[o];
        const float* w = W1 + o * DD;
#pragma unroll
        for (int k = 0; k < DD; ++k) a += w[k] * hp[k];
        y1[o] = a > 0.f ? a : 0.f;
    }
#pragma unroll 4
    for (int o = 0; o < 64; ++o) {
        float a = b2[o];
        const float* w = W2 + o * 64;
#pragma unroll
        for (int k = 0; k < 64; ++k) a += w[k] * y1[k];
        y2[o] = a > 0.f ? a : 0.f;
    }
#pragma unroll
    for (int o = 0; o < 2; ++o) {
        float a = b3[o];
        const float* w = W3 + o * 64;
#pragma unroll
        for (int k = 0; k < 64; ++k) a += w[k] * y2[k];
        a = a > 0.f ? a : 0.f;
        out[n * 2 + o] = 1.f / (1.f + __expf(-a));
    }
}

// ---------------------------------------------------------------------------
extern "C" void kernel_launch(void* const* d_in, const int* in_sizes, int n_in,
                              void* d_out, int out_size, void* d_ws, size_t ws_size,
                              hipStream_t stream) {
    (void)in_sizes; (void)n_in; (void)out_size; (void)ws_size;

    const float* J    = (const float*)d_in[0];
    const float* b    = (const float*)d_in[1];
    const float* mpW1 = (const float*)d_in[2];
    const float* mpb1 = (const float*)d_in[3];
    const float* mpW2 = (const float*)d_in[4];
    const float* mpb2 = (const float*)d_in[5];
    const float* mpW3 = (const float*)d_in[6];
    const float* mpb3 = (const float*)d_in[7];
    const float* Wih  = (const float*)d_in[8];
    const float* Whh  = (const float*)d_in[9];
    const float* bih  = (const float*)d_in[10];
    const float* bhh  = (const float*)d_in[11];
    const float* rW1  = (const float*)d_in[12];
    const float* rb1  = (const float*)d_in[13];
    const float* rW2  = (const float*)d_in[14];
    const float* rb2  = (const float*)d_in[15];
    const float* rW3  = (const float*)d_in[16];
    const float* rb3  = (const float*)d_in[17];

    float* ws  = (float*)d_ws;
    float* hA  = ws;                   // 512*32
    float* hB  = hA + NN * DD;         // 512*32
    float* msg = hB + NN * DD;         // 512*32
    _Float16* W1h = (_Float16*)(msg + NN * DD);   // 64*96 (padded, bias col 67)
    _Float16* W2h = W1h + HM * 96;                // 64*64
    _Float16* W3h = W2h + HM * HM;                // 32*64

    (void)hipFuncSetAttribute((const void*)ggnn_edge,
                              hipFuncAttributeMaxDynamicSharedMemorySize,
                              LDS_TOTAL);

    ggnn_prep<<<64, 256, 0, stream>>>(mpW1, mpb1, mpW2, mpW3,
                                      W1h, W2h, W3h, hA, msg);

    float* hc = hA;
    float* hn = hB;
    for (int s = 0; s < 5; ++s) {
        dim3 grid(NN / 16, NN / 16);
        ggnn_edge<<<grid, 256, LDS_TOTAL, stream>>>(J, b, hc, W1h,
                                                    W2h, mpb2, W3h, mpb3, msg);
        ggnn_gru<<<(NN * DD) / 256, 256, 0, stream>>>(hc, msg, Wih, Whh, bih, bhh, hn);
        float* tmp = hc; hc = hn; hn = tmp;
    }

    ggnn_readout<<<NN / 64, 64, 0, stream>>>(hc, rW1, rb1, rW2, rb2, rW3, rb3,
                                             (float*)d_out);
}
